// FusedSemanticHead_DCN_52209622450328
// MI455X (gfx1250) — compile-verified, matrix operand order checked by reading, not run on hardware
//
#include <hip/hip_runtime.h>
#include <hip/hip_bf16.h>

// ---------------------------------------------------------------------------
// Types for gfx1250 WMMA
// ---------------------------------------------------------------------------
typedef __bf16 bf16_t;
typedef bf16_t v16bf __attribute__((ext_vector_type(16)));
typedef float  v8f   __attribute__((ext_vector_type(8)));
typedef int    v4i   __attribute__((ext_vector_type(4)));
typedef unsigned short ushort;
typedef ushort ushort8 __attribute__((ext_vector_type(8)));

__device__ __forceinline__ ushort f2bf(float v) {
    return (ushort)(__float_as_uint(v) >> 16);
}

#define HW      16384      // 128*128 pixels (M dimension of every GEMM)
#define CDIM    256
#define K33     2304       // 256*9
#define BLK_M   128
#define BLK_N   128
#define KSTEP   32
#define ASTR    48         // halves per LDS row (96B: 32B-aligned fragments)

#define USE_ASYNC_LDS __has_builtin(__builtin_amdgcn_global_load_async_to_lds_b128)

#if USE_ASYNC_LDS
typedef __attribute__((address_space(1))) v4i gas_v4i;   // global
typedef __attribute__((address_space(3))) v4i las_v4i;   // LDS
__device__ __forceinline__ void async_cp16(const void* g, void* l) {
    __builtin_amdgcn_global_load_async_to_lds_b128(
        (gas_v4i*)g, (las_v4i*)l, 0, 0);
}
#endif

// ---------------------------------------------------------------------------
// WMMA implicit-GEMM:  Out[n, m] = sum_k A[m,k] * B[n,k]   (A,B bf16, acc f32)
// Double-buffered LDS staging via GLOBAL_LOAD_ASYNC_TO_LDS_B128 (ASYNCcnt),
// 8 waves (4M x 2N), wave tile 32x64 -> 2x4 v_wmma_f32_16x16x32_bf16 / stage.
// Epilogue fuses BN / bias / ReLU / accumulate; output layout NCHW [N,16384].
// ---------------------------------------------------------------------------
__global__ __launch_bounds__(256)
void wmma_gemm_k(const ushort* __restrict__ A, const ushort* __restrict__ Bw,
                 int K, int N, float* __restrict__ Out,
                 const float* __restrict__ bn_g, const float* __restrict__ bn_b,
                 const float* __restrict__ bn_m, const float* __restrict__ bn_v,
                 const float* __restrict__ bias, int relu, int accum)
{
    __shared__ __attribute__((aligned(32))) ushort As[2][BLK_M * ASTR];
    __shared__ __attribute__((aligned(32))) ushort Bs[2][BLK_N * ASTR];

    const int tid  = threadIdx.x;
    const int lane = tid & 31;
    const int wv   = tid >> 5;
    const int wm   = (wv & 3) * 32;   // wave M offset inside block tile
    const int wn   = (wv >> 2) * 64;  // wave N offset inside block tile
    const int bm   = blockIdx.x * BLK_M;
    const int bnb  = blockIdx.y * BLK_N;

    const v8f vz = {0.f,0.f,0.f,0.f,0.f,0.f,0.f,0.f};
    v8f acc[2][4] = {{vz,vz,vz,vz},{vz,vz,vz,vz}};

    // 2x4 WMMA over one staged 32-K chunk
    auto compute = [&](const ushort* Asb, const ushort* Bsb) {
        v16bf afrag[2];
#pragma unroll
        for (int i = 0; i < 2; ++i) {
            const int r = wm + i * 16 + (lane & 15);
            afrag[i] = *(const v16bf*)&Asb[r * ASTR + (lane >> 4) * 16];
        }
#pragma unroll
        for (int j = 0; j < 4; ++j) {
            const int c = wn + j * 16 + (lane & 15);
            v16bf bfrag = *(const v16bf*)&Bsb[c * ASTR + (lane >> 4) * 16];
#pragma unroll
            for (int i = 0; i < 2; ++i)
                acc[i][j] = __builtin_amdgcn_wmma_f32_16x16x32_bf16(
                    false, afrag[i], false, bfrag, (short)0, acc[i][j],
                    false, false);
        }
    };

#if USE_ASYNC_LDS
    // Each stage: A 128x32 halves (512 x 16B chunks) + B same.
    // chunk -> row = ch>>2, half-offset = (ch&3)*8.  Each thread: 2 A + 2 B
    // async instructions per stage -> each wave issues exactly 4.
    const int c0 = tid * 2;
    auto issue = [&](int buf, int k0) {
#pragma unroll
        for (int u = 0; u < 2; ++u) {
            const int ch  = c0 + u;
            const int row = ch >> 2;
            const int ko  = (ch & 3) * 8;
            async_cp16(A + (size_t)(bm + row) * K + k0 + ko,
                       &As[buf][row * ASTR + ko]);
            const int ncol = min(bnb + row, N - 1);   // clamp: stores guarded later
            async_cp16(Bw + (size_t)ncol * K + k0 + ko,
                       &Bs[buf][row * ASTR + ko]);
        }
    };

    issue(0, 0);
    int buf = 0;
    for (int k0 = 0; k0 < K; k0 += KSTEP) {
        const bool more = (k0 + KSTEP) < K;
        if (more) {
            issue(buf ^ 1, k0 + KSTEP);
            asm volatile("s_wait_asynccnt 0x4" ::: "memory");  // current stage done
        } else {
            asm volatile("s_wait_asynccnt 0x0" ::: "memory");
        }
        __syncthreads();
        compute(&As[buf][0], &Bs[buf][0]);
        __syncthreads();
        buf ^= 1;
    }
#else
    // Synchronous fallback: stage through VGPRs.
    const int row = tid >> 1;
    const int kc  = (tid & 1) * 16;
    for (int k0 = 0; k0 < K; k0 += KSTEP) {
        const ushort* ap = A + (size_t)(bm + row) * K + k0 + kc;
        __builtin_prefetch(ap + KSTEP, 0, 3);           // near-scope prefetch
        ushort8 a0 = *(const ushort8*)ap;
        ushort8 a1 = *(const ushort8*)(ap + 8);
        *(ushort8*)&As[0][row * ASTR + kc]     = a0;
        *(ushort8*)&As[0][row * ASTR + kc + 8] = a1;
        const int ncol = min(bnb + row, N - 1);
        const ushort* bp = Bw + (size_t)ncol * K + k0 + kc;
        ushort8 b0 = *(const ushort8*)bp;
        ushort8 b1 = *(const ushort8*)(bp + 8);
        *(ushort8*)&Bs[0][row * ASTR + kc]     = b0;
        *(ushort8*)&Bs[0][row * ASTR + kc + 8] = b1;
        __syncthreads();
        compute(&As[0][0], &Bs[0][0]);
        __syncthreads();
    }
#endif

    // ---- epilogue: BN / bias / ReLU / accumulate, NCHW store ----
#pragma unroll
    for (int j = 0; j < 4; ++j) {
        const int n = bnb + wn + j * 16 + (lane & 15);
        if (n >= N) continue;
        float s = 1.f, o = 0.f;
        if (bn_g) {
            const float sc = bn_g[n] * rsqrtf(bn_v[n] + 1e-5f);
            s = sc;
            o = bn_b[n] - bn_m[n] * sc;
        }
        if (bias) o += bias[n];
        float* op = Out + (size_t)n * HW;
#pragma unroll
        for (int i = 0; i < 2; ++i) {
            const int mb = bm + wm + i * 16 + (lane >> 4) * 8;
#pragma unroll
            for (int e = 0; e < 8; ++e) {
                float v = acc[i][j][e] * s + o;
                if (relu) v = fmaxf(v, 0.f);
                if (accum) op[mb + e] += v;
                else       op[mb + e] = v;
            }
        }
    }
}

// ---------------------------------------------------------------------------
// Align-corners bilinear resize [256,H,W] -> [256,128,128]
// ---------------------------------------------------------------------------
__global__ __launch_bounds__(256)
void resize_k(const float* __restrict__ in, float* __restrict__ out,
              int H, int W)
{
    const int idx = blockIdx.x * 256 + threadIdx.x;
    if (idx >= CDIM * HW) return;
    const int c  = idx >> 14;
    const int p  = idx & (HW - 1);
    const int oy = p >> 7, ox = p & 127;
    const float sy = (float)(H - 1) / 127.0f;
    const float sx = (float)(W - 1) / 127.0f;
    const float ys = oy * sy, xs = ox * sx;
    int y0 = (int)floorf(ys); y0 = min(max(y0, 0), H - 2);
    int x0 = (int)floorf(xs); x0 = min(max(x0, 0), W - 2);
    const float wy = ys - (float)y0, wx = xs - (float)x0;
    const float* ip = in + (size_t)c * H * W;
    const float a = ip[y0 * W + x0],       b = ip[y0 * W + x0 + 1];
    const float d = ip[(y0 + 1) * W + x0], e = ip[(y0 + 1) * W + x0 + 1];
    out[idx] = (a * (1.f - wy) + d * wy) * (1.f - wx)
             + (b * (1.f - wy) + e * wy) * wx;
}

// ---------------------------------------------------------------------------
// Direct 3x3 conv 256 -> 18 (+bias) for DCN offsets (tiny: ~1.4 GFLOP)
// ---------------------------------------------------------------------------
__global__ __launch_bounds__(256)
void off_conv_k(const float* __restrict__ x, const float* __restrict__ w,
                const float* __restrict__ b, float* __restrict__ o)
{
    const int idx = blockIdx.x * 256 + threadIdx.x;   // co*HW + m
    if (idx >= 18 * HW) return;
    const int co = idx >> 14, m = idx & (HW - 1);
    const int y = m >> 7, xq = m & 127;
    float acc = b[co];
    const float* wp = w + (size_t)co * K33;
    for (int c = 0; c < CDIM; ++c) {
        const float* xp = x + (size_t)c * HW;
        const float* wc = wp + c * 9;
#pragma unroll
        for (int r = 0; r < 9; ++r) {
            const int yy = y + r / 3 - 1, xx = xq + r % 3 - 1;
            if (yy >= 0 && yy < 128 && xx >= 0 && xx < 128)
                acc = fmaf(xp[yy * 128 + xx], wc[r], acc);
        }
    }
    o[idx] = acc;
}

// ---------------------------------------------------------------------------
// Plain 3x3 im2col -> bf16 A[m, c*9+r]
// ---------------------------------------------------------------------------
__global__ __launch_bounds__(256)
void im2col_k(const float* __restrict__ x, ushort* __restrict__ A)
{
    const int idx = blockIdx.x * 256 + threadIdx.x;   // c*HW + m
    if (idx >= CDIM * HW) return;
    const int c = idx >> 14, m = idx & (HW - 1);
    const int y = m >> 7, xq = m & 127;
    const float* xp = x + (size_t)c * HW;
    ushort* ap = A + (size_t)m * K33 + c * 9;
#pragma unroll
    for (int r = 0; r < 9; ++r) {
        const int yy = y + r / 3 - 1, xx = xq + r % 3 - 1;
        const float v = (yy >= 0 && yy < 128 && xx >= 0 && xx < 128)
                        ? xp[yy * 128 + xx] : 0.f;
        ap[r] = f2bf(v);
    }
}

// ---------------------------------------------------------------------------
// Deformable (DCNv1) im2col: bilinear-sample x at offset taps -> bf16 A
// off layout [18, HW]: channel kp*2 = dy, kp*2+1 = dx
// ---------------------------------------------------------------------------
__global__ __launch_bounds__(256)
void dcn_im2col_k(const float* __restrict__ x, const float* __restrict__ off,
                  ushort* __restrict__ A)
{
    const int idx = blockIdx.x * 256 + threadIdx.x;   // m*9 + kp
    if (idx >= HW * 9) return;
    const int m = idx / 9, kp = idx % 9;
    const int y = m >> 7, xq = m & 127;
    const float yy = (float)y + (float)(kp / 3 - 1) + off[(kp * 2 + 0) * HW + m];
    const float xx = (float)xq + (float)(kp % 3 - 1) + off[(kp * 2 + 1) * HW + m];
    const float y0f = floorf(yy), x0f = floorf(xx);
    const int y0 = (int)y0f, x0 = (int)x0f;
    const float wy = yy - y0f, wx = xx - x0f;
    int o00, o01, o10, o11;
    float m00, m01, m10, m11;
    auto pre = [](int yi, int xi, int& o, float& msk) {
        msk = (yi >= 0 && yi <= 127 && xi >= 0 && xi <= 127) ? 1.f : 0.f;
        const int yc = min(max(yi, 0), 127), xc = min(max(xi, 0), 127);
        o = yc * 128 + xc;
    };
    pre(y0,     x0,     o00, m00);
    pre(y0,     x0 + 1, o01, m01);
    pre(y0 + 1, x0,     o10, m10);
    pre(y0 + 1, x0 + 1, o11, m11);
    const float w00 = (1.f - wy) * (1.f - wx) * m00;
    const float w01 = (1.f - wy) * wx * m01;
    const float w10 = wy * (1.f - wx) * m10;
    const float w11 = wy * wx * m11;
    ushort* ap = A + (size_t)m * K33 + kp;
    for (int c = 0; c < CDIM; ++c) {
        const float* xp = x + (size_t)c * HW;
        const float v = xp[o00] * w00 + xp[o01] * w01
                      + xp[o10] * w10 + xp[o11] * w11;
        ap[c * 9] = f2bf(v);
    }
}

// ---------------------------------------------------------------------------
// NCHW fp32 -> bf16 A[m, c] transpose (A operand for 1x1 heads)
// ---------------------------------------------------------------------------
__global__ __launch_bounds__(256)
void cvt_tr_k(const float* __restrict__ x, ushort* __restrict__ A)
{
    const int idx = blockIdx.x * 256 + threadIdx.x;   // c*HW + m
    if (idx >= CDIM * HW) return;
    const int c = idx >> 14, m = idx & (HW - 1);
    A[(size_t)m * CDIM + c] = f2bf(x[idx]);
}

// ---------------------------------------------------------------------------
// fp32 -> bf16 elementwise pack (weights are already [N,K] contiguous)
// ---------------------------------------------------------------------------
__global__ __launch_bounds__(256)
void pack_k(const float* __restrict__ w, ushort* __restrict__ o, int n)
{
    const int i = blockIdx.x * 256 + threadIdx.x;
    if (i < n) o[i] = f2bf(w[i]);
}

// ---------------------------------------------------------------------------
// Host-side launch
// ---------------------------------------------------------------------------
static inline char* ws_take(char*& p, size_t bytes) {
    char* r = p;
    p += (bytes + 255) & ~(size_t)255;
    return r;
}

extern "C" void kernel_launch(void* const* d_in, const int* in_sizes, int n_in,
                              void* d_out, int out_size, void* d_ws, size_t ws_size,
                              hipStream_t stream)
{
    (void)in_sizes; (void)n_in; (void)out_size; (void)ws_size;
    const float* feat[5];
    for (int i = 0; i < 5; ++i) feat[i] = (const float*)d_in[i];
    const float* off1_w = (const float*)d_in[5];
    const float* off1_b = (const float*)d_in[6];
    const float* dc1_w  = (const float*)d_in[7];
    const float* off2_w = (const float*)d_in[8];
    const float* off2_b = (const float*)d_in[9];
    const float* dc2_w  = (const float*)d_in[10];
    const float* bn2_g  = (const float*)d_in[11];
    const float* bn2_b  = (const float*)d_in[12];
    const float* bn2_m  = (const float*)d_in[13];
    const float* bn2_v  = (const float*)d_in[14];
    const float* convs_w = (const float*)d_in[15];   // [4,256,2304]
    const float* convs_g = (const float*)d_in[16];
    const float* convs_b = (const float*)d_in[17];
    const float* convs_m = (const float*)d_in[18];
    const float* convs_v = (const float*)d_in[19];
    const float* emb_w  = (const float*)d_in[20];
    const float* emb_g  = (const float*)d_in[21];
    const float* emb_b  = (const float*)d_in[22];
    const float* emb_m  = (const float*)d_in[23];
    const float* emb_v  = (const float*)d_in[24];
    const float* logit_w = (const float*)d_in[25];
    const float* logit_b = (const float*)d_in[26];

    // ---- carve workspace ----
    char* p = (char*)d_ws;
    ushort* Abuf = (ushort*)ws_take(p, (size_t)HW * K33 * 2);       // im2col A
    ushort* A1   = (ushort*)ws_take(p, (size_t)HW * CDIM * 2);      // 1x1 A
    float*  Rbuf = (float*)ws_take(p, (size_t)CDIM * HW * 4);       // resized feat
    float*  H1   = (float*)ws_take(p, (size_t)CDIM * HW * 4);       // dcn1 out
    float*  SUM  = (float*)ws_take(p, (size_t)CDIM * HW * 4);       // branch sum
    float*  X1   = (float*)ws_take(p, (size_t)CDIM * HW * 4);
    float*  X2   = (float*)ws_take(p, (size_t)CDIM * HW * 4);
    float*  O1   = (float*)ws_take(p, (size_t)18 * HW * 4);
    float*  O2   = (float*)ws_take(p, (size_t)18 * HW * 4);
    ushort* Wdc1 = (ushort*)ws_take(p, (size_t)CDIM * K33 * 2);
    ushort* Wdc2 = (ushort*)ws_take(p, (size_t)CDIM * K33 * 2);
    ushort* Wcv[4];
    for (int i = 0; i < 4; ++i)
        Wcv[i] = (ushort*)ws_take(p, (size_t)CDIM * K33 * 2);
    ushort* Wemb = (ushort*)ws_take(p, (size_t)CDIM * CDIM * 2);
    ushort* Wlog = (ushort*)ws_take(p, (size_t)183 * CDIM * 2);

    const dim3 gblk(256);
    const dim3 gemmGrid(HW / BLK_M, 2);   // N <= 256 everywhere (guarded)
    const int NK33 = CDIM * K33;

    // ---- pack weights to bf16 [N, K] ----
    pack_k<<<(NK33 + 255) / 256, gblk, 0, stream>>>(dc1_w, Wdc1, NK33);
    pack_k<<<(NK33 + 255) / 256, gblk, 0, stream>>>(dc2_w, Wdc2, NK33);
    for (int i = 0; i < 4; ++i)
        pack_k<<<(NK33 + 255) / 256, gblk, 0, stream>>>(convs_w + (size_t)i * NK33,
                                                        Wcv[i], NK33);
    pack_k<<<(CDIM * CDIM + 255) / 256, gblk, 0, stream>>>(emb_w, Wemb, CDIM * CDIM);
    pack_k<<<(183 * CDIM + 255) / 256, gblk, 0, stream>>>(logit_w, Wlog, 183 * CDIM);

    // ---- per-branch pipeline: offsets -> dcn1 -> offsets -> dcn2+BN+ReLU ----
    auto branch = [&](const float* Xin, int accum) {
        off_conv_k<<<(18 * HW + 255) / 256, gblk, 0, stream>>>(Xin, off1_w, off1_b, O1);
        dcn_im2col_k<<<(HW * 9 + 255) / 256, gblk, 0, stream>>>(Xin, O1, Abuf);
        wmma_gemm_k<<<gemmGrid, gblk, 0, stream>>>(Abuf, Wdc1, K33, CDIM, H1,
            nullptr, nullptr, nullptr, nullptr, nullptr, 0, 0);
        off_conv_k<<<(18 * HW + 255) / 256, gblk, 0, stream>>>(H1, off2_w, off2_b, O2);
        dcn_im2col_k<<<(HW * 9 + 255) / 256, gblk, 0, stream>>>(H1, O2, Abuf);
        wmma_gemm_k<<<gemmGrid, gblk, 0, stream>>>(Abuf, Wdc2, K33, CDIM, SUM,
            bn2_g, bn2_b, bn2_m, bn2_v, nullptr, 1, accum);
    };

    // fusion level 1 (native 128x128) first, then the resized branches
    branch(feat[1], 0);
    const int sizes[5] = {256, 128, 64, 32, 16};
    for (int i = 0; i < 5; ++i) {
        if (i == 1) continue;
        resize_k<<<(CDIM * HW + 255) / 256, gblk, 0, stream>>>(feat[i], Rbuf,
                                                               sizes[i], sizes[i]);
        branch(Rbuf, 1);
    }

    // ---- 4x (conv3x3 + BN + ReLU) tower ----
    const float* cur = SUM;
    float* dst[4] = {X1, X2, X1, X2};
    for (int i = 0; i < 4; ++i) {
        im2col_k<<<(CDIM * HW + 255) / 256, gblk, 0, stream>>>(cur, Abuf);
        wmma_gemm_k<<<gemmGrid, gblk, 0, stream>>>(Abuf, Wcv[i], K33, CDIM, dst[i],
            convs_g + i * CDIM, convs_b + i * CDIM,
            convs_m + i * CDIM, convs_v + i * CDIM, nullptr, 1, 0);
        cur = dst[i];
    }

    // ---- heads: mask logits (1x1 + bias) and embedding (1x1 + BN + ReLU) ----
    cvt_tr_k<<<(CDIM * HW + 255) / 256, gblk, 0, stream>>>(cur, A1);
    float* out_mask = (float*)d_out;                 // [183, 16384]
    float* out_emb  = out_mask + (size_t)183 * HW;   // [256, 16384]
    wmma_gemm_k<<<gemmGrid, gblk, 0, stream>>>(A1, Wlog, CDIM, 183, out_mask,
        nullptr, nullptr, nullptr, nullptr, logit_b, 0, 0);
    wmma_gemm_k<<<gemmGrid, gblk, 0, stream>>>(A1, Wemb, CDIM, CDIM, out_emb,
        emb_g, emb_b, emb_m, emb_v, nullptr, 1, 0);
}
